// MBESegNet2_80917183857444
// MI455X (gfx1250) — matrix operand order repaired and checked by reading, not compiled, not run on hardware
//
#include <hip/hip_runtime.h>
#include <hip/hip_bf16.h>

// ---------------------------------------------------------------------------
// MI455X (gfx1250, wave32) implementation.
// Heavy lifting: f16 WMMA GEMM (v_wmma_f32_16x16x32_f16), f32 I/O in HBM,
// branch-free clamped staging, register double-buffering, LDS 64x256 block
// tiles, ds_load_b128 fragment feeds per the CDNA5 ISA A/B VGPR layouts.
// Templated fast paths: KMUL32 (I % 32 == 0) and interior (mnFull) blocks
// skip all tail masking; HASADD folds the residual add.
// ---------------------------------------------------------------------------

typedef __attribute__((ext_vector_type(16))) _Float16 v16h;
typedef __attribute__((ext_vector_type(8)))  _Float16 v8h;
typedef __attribute__((ext_vector_type(8)))  float    v8f;

#define LDA_PAD 40   // halves per row of As (64 x 32 tile): 80B rows, 16B aligned
#define LDB_PAD 40   // halves per row of Bs (256 x 32 tile, stored [n][k])

union HFrag { v16h v; v8h h[2]; };

__device__ __forceinline__ int iminn(int a, int b) { return a < b ? a : b; }

// Branch-free tile loads into registers (indices clamped, masked at store).
template <bool KMUL32>
__device__ __forceinline__ void load_tiles(const float* __restrict__ W,
                                           const float* __restrict__ Xb,
                                           int O, int I, int C,
                                           int o0, int c0, int k0, int tid,
                                           float aR[8], float bR[32])
{
  // A: 64 rows x 32 k, thread -> (row = tid&63, kchunk = (tid>>6)*8)
  {
    int row = tid & 63;
    int kk  = (tid >> 6) * 8;
    int go  = iminn(o0 + row, O - 1);
    const float* wp = W + (size_t)go * I;
    #pragma unroll
    for (int e = 0; e < 8; ++e) {
      int gk = KMUL32 ? (k0 + kk + e) : iminn(k0 + kk + e, I - 1);
      aR[e] = wp[gk];
    }
  }
  // B: 32 k x 256 cols, thread -> col (n = tid), all 32 k (coalesced across lanes)
  {
    int gc = iminn(c0 + tid, C - 1);
    #pragma unroll
    for (int e = 0; e < 32; ++e) {
      int gk = KMUL32 ? (k0 + e) : iminn(k0 + e, I - 1);
      bR[e] = Xb[(size_t)gk * C + gc];
    }
  }
}

// Fast path: full tile, no masking: cvt f32->f16, 128-bit ds stores.
__device__ __forceinline__ void store_tiles_fast(_Float16* As, _Float16* Bs, int tid,
                                                 const float aR[8], const float bR[32])
{
  {
    int row = tid & 63;
    int kk  = (tid >> 6) * 8;
    alignas(16) _Float16 tmp[8];
    #pragma unroll
    for (int e = 0; e < 8; ++e) tmp[e] = (_Float16)aR[e];
    *(v8h*)&As[row * LDA_PAD + kk] = *(const v8h*)tmp;
  }
  {
    alignas(16) _Float16 tmp[32];
    #pragma unroll
    for (int e = 0; e < 32; ++e) tmp[e] = (_Float16)bR[e];
    #pragma unroll
    for (int q = 0; q < 4; ++q)
      *(v8h*)&Bs[tid * LDB_PAD + q * 8] = *(const v8h*)&tmp[q * 8];
  }
}

// Masked path for edge blocks / K tails.
__device__ __forceinline__ void store_tiles_masked(_Float16* As, _Float16* Bs,
                                                   int O, int I, int C,
                                                   int o0, int c0, int k0, int tid,
                                                   const float aR[8], const float bR[32])
{
  {
    int row  = tid & 63;
    int kk   = (tid >> 6) * 8;
    bool rok = (o0 + row) < O;
    alignas(16) _Float16 tmp[8];
    #pragma unroll
    for (int e = 0; e < 8; ++e) {
      bool ok = rok && (k0 + kk + e) < I;
      tmp[e]  = (_Float16)(ok ? aR[e] : 0.f);
    }
    *(v8h*)&As[row * LDA_PAD + kk] = *(const v8h*)tmp;
  }
  {
    bool cok = (c0 + tid) < C;
    alignas(16) _Float16 tmp[32];
    #pragma unroll
    for (int e = 0; e < 32; ++e) {
      bool ok = cok && (k0 + e) < I;
      tmp[e]  = (_Float16)(ok ? bR[e] : 0.f);
    }
    #pragma unroll
    for (int q = 0; q < 4; ++q)
      *(v8h*)&Bs[tid * LDB_PAD + q * 8] = *(const v8h*)&tmp[q * 8];
  }
}

// D(b,O,C) = W(O,I) * X(b,I,C) + bias(O)  [+ addsrc(b,O,C) w/ custom batch stride]
// Block tile 64(M) x 256(N); 8 wave32 waves in a 2x4 grid; 32x64 per wave.
template <bool KMUL32, bool HASADD>
__global__ __launch_bounds__(256) void gemm_bias_kernel(
    const float* __restrict__ W,
    const float* __restrict__ X,
    const float* __restrict__ bias,
    const float* __restrict__ addsrc, long long addBatchStride,
    float* __restrict__ Out,
    int O, int I, int C)
{
  __shared__ alignas(16) _Float16 As[64  * LDA_PAD];
  __shared__ alignas(16) _Float16 Bs[256 * LDB_PAD];

  const int b    = blockIdx.z;
  const int o0   = blockIdx.y * 64;
  const int c0   = blockIdx.x * 256;
  const int tid  = threadIdx.x;
  const int lane = tid & 31;
  const int wave = tid >> 5;
  const int wm   = wave >> 2;   // 0..1  (M)
  const int wn   = wave & 3;    // 0..3  (N)

  // uniform across the block -> scalar branch
  const bool mnFull = (o0 + 64 <= O) && (c0 + 256 <= C);

  const float* __restrict__ Xb = X + (size_t)b * I * C;

  v8f acc[2][4];
  #pragma unroll
  for (int i = 0; i < 2; ++i)
    #pragma unroll
    for (int j = 0; j < 4; ++j) {
      v8f z = {0.f,0.f,0.f,0.f,0.f,0.f,0.f,0.f};
      acc[i][j] = z;
    }

  float aR[8], bR[32];
  load_tiles<KMUL32>(W, Xb, O, I, C, o0, c0, 0, tid, aR, bR);

  for (int k0 = 0; k0 < I; k0 += 32) {
    if (KMUL32 && mnFull)
      store_tiles_fast(As, Bs, tid, aR, bR);
    else
      store_tiles_masked(As, Bs, O, I, C, o0, c0, k0, tid, aR, bR);
    __syncthreads();

    // prefetch next K tile into registers; overlaps with the WMMA burst below
    if (k0 + 32 < I) {
      load_tiles<KMUL32>(W, Xb, O, I, C, o0, c0, k0 + 32, tid, aR, bR);
    }

    // Fragment loads per CDNA5 16-bit A/B layouts (CDNA5 ISA §7.12.2):
    // A 16x32: lanes 0-15 K=0..7/16..23, lanes 16-31 K=8..15/24..31
    // B 32x16 (LDS stored [n][k]): lanes 0-15 K=0..7/16..23, lanes 16-31 K=8..15/24..31
    const int kA   = (lane & 16) ? 8  : 0;
    const int kB   = (lane & 16) ? 8  : 0;
    const int mrow = wm * 32 + (lane & 15);
    const int nrow = wn * 64 + (lane & 15);

    HFrag a[2], bt[4];
    #pragma unroll
    for (int t = 0; t < 2; ++t) {
      a[t].h[0] = *(const v8h*)&As[(mrow + t * 16) * LDA_PAD + kA];
      a[t].h[1] = *(const v8h*)&As[(mrow + t * 16) * LDA_PAD + kA + 16];
    }
    #pragma unroll
    for (int t = 0; t < 4; ++t) {
      bt[t].h[0] = *(const v8h*)&Bs[(nrow + t * 16) * LDB_PAD + kB];
      bt[t].h[1] = *(const v8h*)&Bs[(nrow + t * 16) * LDB_PAD + kB + 16];
    }
    #pragma unroll
    for (int ms = 0; ms < 2; ++ms)
      #pragma unroll
      for (int ns = 0; ns < 4; ++ns)
        acc[ms][ns] = __builtin_amdgcn_wmma_f32_16x16x32_f16(
            false, a[ms].v, false, bt[ns].v, (short)0, acc[ms][ns], false, false);
    __syncthreads();
  }

  // Epilogue. C/D layout: VGPR r -> M = r + (lane>=16 ? 8 : 0), N = lane&15
  const int rofs = (lane & 16) ? 8 : 0;
  const float* addb = addsrc + (size_t)b * addBatchStride;
  float* outb = Out + (size_t)b * O * C;

  if (mnFull) {
    #pragma unroll
    for (int ms = 0; ms < 2; ++ms) {
      #pragma unroll
      for (int ns = 0; ns < 4; ++ns) {
        int col = c0 + wn * 64 + ns * 16 + (lane & 15);
        #pragma unroll
        for (int r = 0; r < 8; ++r) {
          int gr = o0 + wm * 32 + ms * 16 + rofs + r;
          float v = acc[ms][ns][r] + bias[gr];
          size_t off = (size_t)gr * C + col;
          if (HASADD) v += addb[off];
          outb[off] = v;
        }
      }
    }
  } else {
    #pragma unroll
    for (int ms = 0; ms < 2; ++ms) {
      #pragma unroll
      for (int ns = 0; ns < 4; ++ns) {
        int col = c0 + wn * 64 + ns * 16 + (lane & 15);
        if (col >= C) continue;
        #pragma unroll
        for (int r = 0; r < 8; ++r) {
          int gr = o0 + wm * 32 + ms * 16 + rofs + r;
          if (gr >= O) continue;
          float v = acc[ms][ns][r] + bias[gr];
          size_t off = (size_t)gr * C + col;
          if (HASADD) v += addb[off];
          outb[off] = v;
        }
      }
    }
  }
}

// ---------------------------------------------------------------------------
// Brute-force KNN on centers x[:,9:12,:] keeping top-K of pd = -||xi-xj||^2
// ---------------------------------------------------------------------------
template <int K>
__global__ __launch_bounds__(256) void knn_kernel(const float* __restrict__ x,
                                                  int* __restrict__ idx, int N)
{
  __shared__ float tx[256], ty[256], tz[256];
  const int b = blockIdx.y;
  const int i = blockIdx.x * 256 + threadIdx.x;
  const float* cx = x + (size_t)b * 15 * N + (size_t)9 * N;

  float px = 0.f, py = 0.f, pz = 0.f;
  if (i < N) { px = cx[i]; py = cx[N + i]; pz = cx[2 * N + i]; }

  float bestd[K]; int besti[K];
  #pragma unroll
  for (int t = 0; t < K; ++t) { bestd[t] = -3.4e38f; besti[t] = 0; }

  for (int j0 = 0; j0 < N; j0 += 256) {
    int j = j0 + threadIdx.x;
    tx[threadIdx.x] = (j < N) ? cx[j]         : 3.4e38f;
    ty[threadIdx.x] = (j < N) ? cx[N + j]     : 3.4e38f;
    tz[threadIdx.x] = (j < N) ? cx[2 * N + j] : 3.4e38f;
    __syncthreads();
    int lim = (N - j0 < 256) ? (N - j0) : 256;
    for (int jj = 0; jj < lim; ++jj) {
      float dx = px - tx[jj], dy = py - ty[jj], dz = pz - tz[jj];
      float pd = -(dx * dx + dy * dy + dz * dz);
      if (pd > bestd[K - 1]) {
        int p = K - 1;
        while (p > 0 && bestd[p - 1] < pd) {
          bestd[p] = bestd[p - 1]; besti[p] = besti[p - 1]; --p;
        }
        bestd[p] = pd; besti[p] = j0 + jj;
      }
    }
    __syncthreads();
  }
  if (i < N)
    for (int t = 0; t < K; ++t) idx[((size_t)b * N + i) * K + t] = besti[t];
}

// ---------------------------------------------------------------------------
// BEM input builder: cat1=[tile_xyz;neigh_xyz](18), cat2=[tile_f;neigh_f](2C),
// cat3=[neigh_f;tile_f-neigh_f](2C), cat4=[neigh_xyz;tile_xyz-neigh_xyz](18)
// tile columns use j = kk*N+i (i = j%N); neigh columns use j = i*K+kk.
// ---------------------------------------------------------------------------
__global__ void bem_gather_kernel(const float* __restrict__ x,
                                  const float* __restrict__ feat,
                                  const int* __restrict__ idx,
                                  float* __restrict__ cat1, float* __restrict__ cat2,
                                  float* __restrict__ cat3, float* __restrict__ cat4,
                                  int N, int K, int C)
{
  const int b = blockIdx.y;
  const long long KN = (long long)K * N;
  long long j = (long long)blockIdx.x * 256 + threadIdx.x;
  if (j >= KN) return;
  const int i_t  = (int)(j % N);
  const int i_n  = (int)(j / K);
  const int kk_n = (int)(j % K);
  const int nb   = idx[((size_t)b * N + i_n) * K + kk_n];

  const float* xyzb = x + (size_t)b * 15 * N;          // first 9N floats, (n,9) view
  const float* fb   = feat + (size_t)b * C * N;        // (C, n)

  #pragma unroll
  for (int c = 0; c < 9; ++c) {
    float tv = xyzb[(size_t)i_t * 9 + c];
    float nv = xyzb[(size_t)nb  * 9 + c];
    cat1[((size_t)b * 18 + c)     * KN + j] = tv;
    cat1[((size_t)b * 18 + 9 + c) * KN + j] = nv;
    cat4[((size_t)b * 18 + c)     * KN + j] = nv;
    cat4[((size_t)b * 18 + 9 + c) * KN + j] = tv - nv;
  }
  for (int c = 0; c < C; ++c) {
    float tf = fb[(size_t)c * N + i_t];
    float nf = fb[(size_t)c * N + nb];
    cat2[((size_t)b * 2 * C + c)     * KN + j] = tf;
    cat2[((size_t)b * 2 * C + C + c) * KN + j] = nf;
    cat3[((size_t)b * 2 * C + c)     * KN + j] = nf;
    cat3[((size_t)b * 2 * C + C + c) * KN + j] = tf - nf;
  }
}

// generic channel concat: dst(b, ca+cb, cols)
__global__ void concat2_kernel(const float* __restrict__ A, int ca,
                               const float* __restrict__ Bs, int cb,
                               float* __restrict__ dst, long long cols, long long total)
{
  long long e = (long long)blockIdx.x * 256 + threadIdx.x;
  if (e >= total) return;
  long long percol = (long long)(ca + cb) * cols;
  int b  = (int)(e / percol);
  long long r = e % percol;
  int ch = (int)(r / cols);
  long long j = r % cols;
  float v = (ch < ca) ? A[((size_t)b * ca + ch) * cols + j]
                      : Bs[((size_t)b * cb + (ch - ca)) * cols + j];
  dst[e] = v;
}

// fused: attn = softmax(conv7, over OC within (b*K+kk, col)); api = sum_k attn*lpi;
// afi = max_k lfi; si = concat(api, afi) -> (b, 2*OC, n). blockDim = OC.
__global__ void attn_fuse_kernel(const float* __restrict__ a7,
                                 const float* __restrict__ lpi,
                                 const float* __restrict__ lfi,
                                 float* __restrict__ si,
                                 int Nn, int K, int OC)
{
  extern __shared__ float red[];
  const int col = blockIdx.x;
  const int b   = blockIdx.y;
  const int ch  = threadIdx.x;

  float api = 0.f, afi = -3.4e38f;
  for (int kk = 0; kk < K; ++kk) {
    size_t base = ((size_t)(b * K + kk) * OC + ch) * Nn + col;
    float v7 = a7[base];
    red[ch] = v7; __syncthreads();
    for (int s = OC >> 1; s > 0; s >>= 1) {
      if (ch < s) red[ch] = fmaxf(red[ch], red[ch + s]);
      __syncthreads();
    }
    float m = red[0]; __syncthreads();
    float e = __expf(v7 - m);
    red[ch] = e; __syncthreads();
    for (int s = OC >> 1; s > 0; s >>= 1) {
      if (ch < s) red[ch] += red[ch + s];
      __syncthreads();
    }
    float sum = red[0]; __syncthreads();
    api += (e / sum) * lpi[base];
    afi  = fmaxf(afi, lfi[base]);
  }
  si[((size_t)b * 2 * OC + ch)      * Nn + col] = api;
  si[((size_t)b * 2 * OC + OC + ch) * Nn + col] = afi;
}

// per-channel max over columns
__global__ void max_cols_kernel(const float* __restrict__ X, float* __restrict__ outp,
                                int Cch, int Nn, int outStride, int ch0)
{
  __shared__ float sm[256];
  const int ch = blockIdx.x, b = blockIdx.y, tid = threadIdx.x;
  const float* p = X + ((size_t)b * Cch + ch) * Nn;
  float m = -3.4e38f;
  for (int i = tid; i < Nn; i += 256) m = fmaxf(m, p[i]);
  sm[tid] = m; __syncthreads();
  for (int s = 128; s > 0; s >>= 1) {
    if (tid < s) sm[tid] = fmaxf(sm[tid], sm[tid + s]);
    __syncthreads();
  }
  if (tid == 0) outp[(size_t)b * outStride + ch0 + ch] = sm[0];
}

// h(b,1088,n) = concat(si1(64), si2(256), si3(256), up(512 broadcast))
__global__ void build_h_kernel(const float* __restrict__ si1, const float* __restrict__ si2,
                               const float* __restrict__ si3, const float* __restrict__ up,
                               float* __restrict__ h, int Nn, long long total)
{
  long long e = (long long)blockIdx.x * 256 + threadIdx.x;
  if (e >= total) return;
  int col = (int)(e % Nn);
  long long r = e / Nn;
  int ch = (int)(r % 1088);
  int b  = (int)(r / 1088);
  float v;
  if      (ch <  64) v = si1[((size_t)b * 64  +  ch       ) * Nn + col];
  else if (ch < 320) v = si2[((size_t)b * 256 + (ch - 64 )) * Nn + col];
  else if (ch < 576) v = si3[((size_t)b * 256 + (ch - 320)) * Nn + col];
  else               v = up[(size_t)b * 512 + (ch - 576)];
  h[e] = v;
}

__global__ void bn_stats_kernel(const float* __restrict__ X, float* __restrict__ mean,
                                float* __restrict__ var, int B, int Cch, int Nn)
{
  __shared__ float s1[256], s2[256];
  const int ch = blockIdx.x, tid = threadIdx.x;
  float sum = 0.f, sq = 0.f;
  for (int b = 0; b < B; ++b) {
    const float* p = X + ((size_t)b * Cch + ch) * Nn;
    for (int i = tid; i < Nn; i += 256) { float v = p[i]; sum += v; sq += v * v; }
  }
  s1[tid] = sum; s2[tid] = sq; __syncthreads();
  for (int s = 128; s > 0; s >>= 1) {
    if (tid < s) { s1[tid] += s1[tid + s]; s2[tid] += s2[tid + s]; }
    __syncthreads();
  }
  if (tid == 0) {
    float inv = 1.f / (float)(B * Nn);
    float m = s1[0] * inv;
    mean[ch] = m;
    var[ch]  = s2[0] * inv - m * m;
  }
}

__global__ void bn_relu_kernel(float* __restrict__ Xm, const float* __restrict__ mean,
                               const float* __restrict__ var, const float* __restrict__ g,
                               const float* __restrict__ beta, int Cch, int Nn, long long total)
{
  long long e = (long long)blockIdx.x * 256 + threadIdx.x;
  if (e >= total) return;
  int ch = (int)((e / Nn) % Cch);
  float v = (Xm[e] - mean[ch]) * rsqrtf(var[ch] + 1e-5f) * g[ch] + beta[ch];
  Xm[e] = v > 0.f ? v : 0.f;
}

// probs(b,n,15) = softmax over channels of logits(b,15,n)
__global__ void softmax_out_kernel(const float* __restrict__ logits,
                                   float* __restrict__ probs, int B, int Nn)
{
  long long t = (long long)blockIdx.x * 256 + threadIdx.x;
  if (t >= (long long)B * Nn) return;
  int b = (int)(t / Nn), col = (int)(t % Nn);
  const float* p = logits + (size_t)b * 15 * Nn;
  float m = -3.4e38f;
  #pragma unroll
  for (int c = 0; c < 15; ++c) m = fmaxf(m, p[(size_t)c * Nn + col]);
  float e[15]; float s = 0.f;
  #pragma unroll
  for (int c = 0; c < 15; ++c) { e[c] = __expf(p[(size_t)c * Nn + col] - m); s += e[c]; }
  float inv = 1.f / s;
  #pragma unroll
  for (int c = 0; c < 15; ++c) probs[((size_t)b * Nn + col) * 15 + c] = e[c] * inv;
}

__global__ void coords_copy_kernel(const float* __restrict__ x, float* __restrict__ outc,
                                   int Nn, long long total)
{
  long long e = (long long)blockIdx.x * 256 + threadIdx.x;
  if (e >= total) return;
  int b = (int)(e / (9 * (long long)Nn));
  long long r = e % (9 * (long long)Nn);
  outc[e] = x[(size_t)b * 15 * Nn + r];
}

// ---------------------------------------------------------------------------
extern "C" void kernel_launch(void* const* d_in, const int* in_sizes, int n_in,
                              void* d_out, int out_size, void* d_ws, size_t ws_size,
                              hipStream_t stream)
{
  (void)in_sizes; (void)out_size; (void)ws_size;
  if (n_in < 59) return;

  const int B = 2, N = 10000;
  const float* x = (const float*)d_in[0];
  auto F = [&](int i) { return (const float*)d_in[i]; };

  const float* fea_w = F(1); const float* fea_b = F(2);
  const float* bw[3][7]; const float* bb[3][7];
  {
    int p = 3;
    for (int s = 0; s < 3; ++s)
      for (int m = 0; m < 7; ++m) { bw[s][m] = F(p++); bb[s][m] = F(p++); }
  }
  const float* mlpw[3] = { F(45), F(47), F(49) };
  const float* mlpb[3] = { F(46), F(48), F(50) };
  const float* bng[3]  = { F(51), F(53), F(55) };
  const float* bnb[3]  = { F(52), F(54), F(56) };
  const float* outw = F(57); const float* outb = F(58);

  // ---- output layout: probs | s1 | s2 | s3 | coords
  float* out_probs  = (float*)d_out;                              // B*N*15
  float* out_s1     = out_probs + (size_t)B * N * 15;             // B*9*4N
  float* out_s2     = out_s1 + (size_t)B * 9 * 4 * N;             // B*9*4N
  float* out_s3     = out_s2 + (size_t)B * 9 * 4 * N;             // B*9*16N
  float* out_coords = out_s3 + (size_t)B * 9 * 16 * N;            // B*9*N

  // ---- workspace bump allocator
  char* base = (char*)d_ws; size_t off = 0;
  auto allocF = [&](size_t n) { float* p = (float*)(base + off);
                                off = (off + n * sizeof(float) + 255) & ~(size_t)255; return p; };
  auto allocI = [&](size_t n) { int* p = (int*)(base + off);
                                off = (off + n * sizeof(int) + 255) & ~(size_t)255; return p; };

  auto gemm = [&](const float* W, const float* Xin, const float* bias, float* Out,
                  int O, int I, long long C, const float* addsrc, long long addStride) {
    dim3 g((unsigned)((C + 255) / 256), (unsigned)((O + 63) / 64), B);
    bool k32 = (I % 32) == 0;
    if (addsrc) {
      if (k32) gemm_bias_kernel<true , true ><<<g, 256, 0, stream>>>(W, Xin, bias, addsrc, addStride, Out, O, I, (int)C);
      else     gemm_bias_kernel<false, true ><<<g, 256, 0, stream>>>(W, Xin, bias, addsrc, addStride, Out, O, I, (int)C);
    } else {
      if (k32) gemm_bias_kernel<true , false><<<g, 256, 0, stream>>>(W, Xin, bias, Out, 0, Out, O, I, (int)C);
      else     gemm_bias_kernel<false, false><<<g, 256, 0, stream>>>(W, Xin, bias, Out, 0, Out, O, I, (int)C);
    }
  };

  // ---- persistent buffers
  float* feas = allocF((size_t)B * 16 * N);
  int*   idx1 = allocI((size_t)B * N * 4);
  int*   idx2 = allocI((size_t)B * N * 16);
  float* si1  = allocF((size_t)B * 64 * N);
  float* si2  = allocF((size_t)B * 256 * N);
  float* si3  = allocF((size_t)B * 256 * N);
  float* h    = allocF((size_t)B * 1088 * N);
  float* h1   = allocF((size_t)B * 512 * N);
  float* h2   = allocF((size_t)B * 256 * N);
  float* h3   = allocF((size_t)B * 128 * N);
  float* lgt  = allocF((size_t)B * 15 * N);
  float* up   = allocF((size_t)B * 512);
  float* mean = allocF(512);
  float* var  = allocF(512);
  const size_t arenaBase = off;

  // ---- fea conv + knn
  gemm(fea_w, x, fea_b, feas, 16, 15, N, nullptr, 0);
  knn_kernel<4><<<dim3((N + 255) / 256, B), 256, 0, stream>>>(x, idx1, N);
  knn_kernel<16><<<dim3((N + 255) / 256, B), 256, 0, stream>>>(x, idx2, N);

  // ---- BEM stage (shared arena; reset per stage)
  auto bem = [&](int s, const float* feat, int Cf, const int* idx, int K, int OC,
                 float* si, float* sx_out) {
    off = arenaBase;
    const long long KN = (long long)K * N;
    float* cat1 = allocF((size_t)B * 18 * KN);
    float* cat2 = allocF((size_t)B * 2 * Cf * KN);
    float* cat3 = allocF((size_t)B * 2 * Cf * KN);
    float* cat4 = allocF((size_t)B * 18 * KN);
    float* pij  = allocF((size_t)B * 18 * KN);
    float* fij  = allocF((size_t)B * 2 * Cf * KN);
    float* sf   = allocF((size_t)B * Cf * KN);
    float* cat6 = allocF((size_t)B * 3 * Cf * KN);
    float* lpi  = allocF((size_t)B * OC * KN);
    float* lfi  = allocF((size_t)B * OC * KN);
    float* cat5 = cat3;  // cat3 dead before cat5 is built (27 <= 2*Cf channels)
    float* a7   = cat2;  // cat2 dead after sf residual   (OC == 2*Cf here)

    bem_gather_kernel<<<dim3((unsigned)((KN + 255) / 256), B), 256, 0, stream>>>(
        x, feat, idx, cat1, cat2, cat3, cat4, N, K, Cf);

    gemm(bw[s][0], cat1, bb[s][0], pij, 18,     18,     KN, nullptr, 0);
    gemm(bw[s][1], cat2, bb[s][1], fij, 2 * Cf, 2 * Cf, KN, nullptr, 0);
    gemm(bw[s][2], cat3, bb[s][2], sx_out, 9,  2 * Cf, KN, cat1 + 9 * KN,  18LL * KN);
    gemm(bw[s][3], cat4, bb[s][3], sf, Cf,     18,     KN, cat2 + Cf * KN, 2LL * Cf * KN);

    {
      long long t5 = (long long)B * 27 * KN;
      concat2_kernel<<<(unsigned)((t5 + 255) / 256), 256, 0, stream>>>(sx_out, 9, pij, 18, cat5, KN, t5);
      long long t6 = (long long)B * 3 * Cf * KN;
      concat2_kernel<<<(unsigned)((t6 + 255) / 256), 256, 0, stream>>>(sf, Cf, fij, 2 * Cf, cat6, KN, t6);
    }
    gemm(bw[s][4], cat5, bb[s][4], lpi, OC, 27,     KN, nullptr, 0);
    gemm(bw[s][5], cat6, bb[s][5], lfi, OC, 3 * Cf, KN, nullptr, 0);
    gemm(bw[s][6], lfi,  bb[s][6], a7,  OC, OC,     KN, nullptr, 0);

    attn_fuse_kernel<<<dim3(N, B), OC, OC * sizeof(float), stream>>>(a7, lpi, lfi, si, N, K, OC);
  };

  bem(0, feas, 16, idx1, 4,  32,  si1, out_s1);
  bem(1, si1,  64, idx1, 4,  128, si2, out_s2);
  bem(2, si1,  64, idx2, 16, 128, si3, out_s3);

  // ---- up = max over cols of comb = [si2; si3]
  max_cols_kernel<<<dim3(256, B), 256, 0, stream>>>(si2, up, 256, N, 512, 0);
  max_cols_kernel<<<dim3(256, B), 256, 0, stream>>>(si3, up, 256, N, 512, 256);

  {
    long long th = (long long)B * 1088 * N;
    build_h_kernel<<<(unsigned)((th + 255) / 256), 256, 0, stream>>>(si1, si2, si3, up, h, N, th);
  }

  // ---- MLP head: gemm -> batch-norm -> relu
  auto mlp = [&](int s, const float* in, float* outp, int O, int I) {
    gemm(mlpw[s], in, mlpb[s], outp, O, I, N, nullptr, 0);
    bn_stats_kernel<<<O, 256, 0, stream>>>(outp, mean, var, B, O, N);
    long long t = (long long)B * O * N;
    bn_relu_kernel<<<(unsigned)((t + 255) / 256), 256, 0, stream>>>(outp, mean, var, bng[s], bnb[s], O, N, t);
  };
  mlp(0, h,  h1, 512, 1088);
  mlp(1, h1, h2, 256, 512);
  mlp(2, h2, h3, 128, 256);

  gemm(outw, h3, outb, lgt, 15, 128, N, nullptr, 0);
  softmax_out_kernel<<<(unsigned)(((long long)B * N + 255) / 256), 256, 0, stream>>>(lgt, out_probs, B, N);

  {
    long long tc = (long long)B * 9 * N;
    coords_copy_kernel<<<(unsigned)((tc + 255) / 256), 256, 0, stream>>>(x, out_coords, N, tc);
  }
}
